// transformer_27882927686144
// MI455X (gfx1250) — matrix operand order repaired and checked by reading, not compiled, not run on hardware
//
#include <hip/hip_runtime.h>
#include <hip/hip_bf16.h>
#include <math.h>

typedef __attribute__((ext_vector_type(16))) _Float16 v16h;
typedef __attribute__((ext_vector_type(8)))  _Float16 v8h;
typedef __attribute__((ext_vector_type(8)))  float    v8f;
typedef __attribute__((ext_vector_type(4)))  int      v4i;

#define AS1 __attribute__((address_space(1)))
#define AS3 __attribute__((address_space(3)))

#if defined(__has_builtin)
#if __has_builtin(__builtin_amdgcn_global_load_async_to_lds_b128) && \
    __has_builtin(__builtin_amdgcn_s_wait_asynccnt)
#define USE_ASYNC_LDS 1
#endif
#endif
#ifndef USE_ASYNC_LDS
#define USE_ASYNC_LDS 0
#endif

#if USE_ASYNC_LDS
#define ASYNC_WAIT(n) __builtin_amdgcn_s_wait_asynccnt(n)
#else
#define ASYNC_WAIT(n)
#endif

// 16-byte global -> LDS copy: async (GLOBAL_LOAD_ASYNC_TO_LDS_B128) when available.
__device__ __forceinline__ void cp16_g2l(const _Float16* g, _Float16* l) {
#if USE_ASYNC_LDS
  __builtin_amdgcn_global_load_async_to_lds_b128((AS1 v4i*)g, (AS3 v4i*)l, 0, 0);
#else
  *(uint4*)l = *(const uint4*)g;
#endif
}

__device__ __forceinline__ v16h cat8(v8h lo, v8h hi) {
  return __builtin_shufflevector(lo, hi, 0,1,2,3,4,5,6,7,8,9,10,11,12,13,14,15);
}
__device__ __forceinline__ v8f wmma16(v16h a, v16h b, v8f c) {
  return __builtin_amdgcn_wmma_f32_16x16x32_f16(false, a, false, b, (short)0, c, false, false);
}

// ---------------- elementwise / conversion kernels ----------------

__global__ void f32_to_f16_kernel(const float* __restrict__ s, _Float16* __restrict__ d, size_t n) {
  size_t i = (size_t)blockIdx.x * blockDim.x + threadIdx.x;
  if (i < n) d[i] = (_Float16)s[i];
}

// distil conv weight: (o, i, d) -> (o, d*512 + i) as f16
__global__ void conv_w_kernel(const float* __restrict__ s, _Float16* __restrict__ d) {
  int i = blockIdx.x * blockDim.x + threadIdx.x;
  if (i >= 512 * 1536) return;
  int n = i / 1536, k = i % 1536, dd = k >> 9, ci = k & 511;
  d[i] = (_Float16)s[n * 1536 + ci * 3 + dd];
}

// token conv (wrap pad, k=3, 8 in-ch) + sinusoidal pos embedding -> X f32 + Xh f16
__global__ __launch_bounds__(128) void tok_pos_kernel(
    const float* __restrict__ xe, const float* __restrict__ tw, const float* __restrict__ tb,
    float* __restrict__ X, _Float16* __restrict__ Xh, int L) {
  int row = blockIdx.x;           // b*L + l
  int b = row / L, l = row % L;
  int t = threadIdx.x;            // 128 threads, 4 channels each
  __shared__ float xs[24];
  if (t < 24) {
    int dd = t / 8, i = t % 8;
    int ll = (l - 1 + dd + L) % L;
    xs[t] = xe[((size_t)b * L + ll) * 8 + i];
  }
  __syncthreads();
  for (int j = 0; j < 4; ++j) {
    int o = t * 4 + j;
    float acc = tb[o];
    const float* w = tw + o * 24;     // (o, i, d)
    #pragma unroll
    for (int i = 0; i < 8; ++i)
      #pragma unroll
      for (int dd = 0; dd < 3; ++dd)
        acc += w[i * 3 + dd] * xs[dd * 8 + i];
    float div = __expf(-9.210340371976184f * (float)(o & ~1) * (1.0f / 512.0f));
    float ang = (float)l * div;
    acc += (o & 1) ? cosf(ang) : sinf(ang);
    X [(size_t)row * 512 + o] = acc;
    Xh[(size_t)row * 512 + o] = (_Float16)acc;
  }
}

// layernorm( Xi [+ R] ) -> Xo f32 (+ Ho f16), D=512, one row per 128-thread block
__global__ __launch_bounds__(128) void ln_kernel(
    const float* __restrict__ Xi, const float* __restrict__ R,
    const float* __restrict__ gg, const float* __restrict__ bb,
    float* __restrict__ Xo, _Float16* __restrict__ Ho) {
  __shared__ float red[128];
  int row = blockIdx.x, t = threadIdx.x;
  const float* xr = Xi + (size_t)row * 512;
  const float* rr = R ? R + (size_t)row * 512 : nullptr;
  float v[4]; float s = 0.f;
  #pragma unroll
  for (int i = 0; i < 4; ++i) {
    int c = t * 4 + i;
    v[i] = xr[c] + (rr ? rr[c] : 0.f);
    s += v[i];
  }
  red[t] = s; __syncthreads();
  for (int o = 64; o > 0; o >>= 1) { if (t < o) red[t] += red[t + o]; __syncthreads(); }
  float mean = red[0] * (1.f / 512.f);
  __syncthreads();
  s = 0.f;
  #pragma unroll
  for (int i = 0; i < 4; ++i) { float d = v[i] - mean; s += d * d; }
  red[t] = s; __syncthreads();
  for (int o = 64; o > 0; o >>= 1) { if (t < o) red[t] += red[t + o]; __syncthreads(); }
  float inv = rsqrtf(red[0] * (1.f / 512.f) + 1e-5f);
  #pragma unroll
  for (int i = 0; i < 4; ++i) {
    int c = t * 4 + i;
    float o2 = (v[i] - mean) * inv * gg[c] + bb[c];
    Xo[(size_t)row * 512 + c] = o2;
    if (Ho) Ho[(size_t)row * 512 + c] = (_Float16)o2;
  }
}

// im2col for distil conv: row (b,j), K index d*512+i <- Xh[b, (j-2+d) mod L, i]
__global__ __launch_bounds__(256) void im2col_kernel(
    const _Float16* __restrict__ Xh, _Float16* __restrict__ XC, int L, int Lc) {
  int row = blockIdx.x;            // b*Lc + j
  int b = row / Lc, j = row % Lc;
  int t = threadIdx.x;
  for (int s = 0; s < 6; ++s) {
    int col = t + s * 256;
    int dd = col >> 9, i = col & 511;
    int l = j - 2 + dd;
    if (l < 0)  l += L;
    if (l >= L) l -= L;
    XC[(size_t)row * 1536 + col] = Xh[((size_t)b * L + l) * 512 + i];
  }
}

// scale/shift + elu + maxpool(3, stride 2, pad 1) -> new X/Xh
__global__ __launch_bounds__(256) void distil_post_kernel(
    const float* __restrict__ Yc, const float* __restrict__ gg, const float* __restrict__ bb,
    float* __restrict__ X, _Float16* __restrict__ Xh, int Lc, int Lo) {
  int row = blockIdx.x;            // b*Lo + lo
  int b = row / Lo, lo = row % Lo;
  int t = threadIdx.x;
  const float inv = 0.99999500003749969f;   // 1/sqrt(1+1e-5)
  for (int s = 0; s < 2; ++s) {
    int c = t + s * 256;
    float mx = -3.0e38f;
    #pragma unroll
    for (int w = 0; w < 3; ++w) {
      int j = 2 * lo - 1 + w;
      if (j < 0 || j >= Lc) continue;
      float y = Yc[((size_t)b * Lc + j) * 512 + c];
      y = y * inv * gg[c] + bb[c];
      y = (y > 0.f) ? y : (__expf(y) - 1.0f);
      mx = fmaxf(mx, y);
    }
    X [(size_t)row * 512 + c] = mx;
    Xh[(size_t)row * 512 + c] = (_Float16)mx;
  }
}

__global__ void bnelu_kernel(const float* __restrict__ H, const float* __restrict__ gg,
                             const float* __restrict__ bb, _Float16* __restrict__ O, int n, int D) {
  int i = blockIdx.x * blockDim.x + threadIdx.x;
  if (i >= n) return;
  int c = i % D;
  float y = H[i] * 0.99999500003749969f * gg[c] + bb[c];
  y = (y > 0.f) ? y : (__expf(y) - 1.0f);
  O[i] = (_Float16)y;
}

__global__ void take_last_kernel(const _Float16* __restrict__ Xh, _Float16* __restrict__ out, int L) {
  int i = blockIdx.x * blockDim.x + threadIdx.x;   // 8*512
  if (i >= 8 * 512) return;
  int b = i >> 9, c = i & 511;
  out[i] = Xh[((size_t)b * L + (L - 1)) * 512 + c];
}

// ---------------- WMMA GEMM:  C[M,N] = act(A[M,K] @ W[N,K]^T + bias) ----------------
// 64x64 block tile, BK=64, 8 waves (4Mx2N), each wave 16x32 -> 4 WMMAs per stage.
// Double-buffered LDS filled by async global->LDS (ASYNCcnt) when available.
// Tail M rows are clamped (not zeroed): garbage A rows only affect masked D rows.
__global__ __launch_bounds__(256) void gemm_kernel(
    const _Float16* __restrict__ A, const _Float16* __restrict__ W,
    const float* __restrict__ bias, float* __restrict__ Cf, _Float16* __restrict__ Ch,
    int M, int N, int K, int act) {
  __shared__ __align__(128) _Float16 As[2][64 * 64];
  __shared__ __align__(128) _Float16 Bs[2][64 * 64];
  int t = threadIdx.x, lane = t & 31, wave = t >> 5;
  int wm = wave >> 1, wn = wave & 1;
  int m0 = blockIdx.x * 64, n0 = blockIdx.y * 64;
  int g = lane >> 4, ln = lane & 15;
  v8f c0 = {0,0,0,0,0,0,0,0}, c1 = {0,0,0,0,0,0,0,0};

  // tile fill: 64x64 halves per matrix = 512 b128 chunks; each thread moves 2 per matrix
  int r0 = t >> 3, col8 = (t & 7) * 8;
  int r1 = r0 + 32;
  int mA0 = m0 + r0; if (mA0 >= M) mA0 = M - 1;
  int mA1 = m0 + r1; if (mA1 >= M) mA1 = M - 1;
  const _Float16* a0 = A + (size_t)mA0 * K + col8;
  const _Float16* a1 = A + (size_t)mA1 * K + col8;
  const _Float16* w0 = W + (size_t)(n0 + r0) * K + col8;
  const _Float16* w1 = W + (size_t)(n0 + r1) * K + col8;

  auto stage = [&](int buf, int k0) {
    cp16_g2l(a0 + k0, &As[buf][r0 * 64 + col8]);
    cp16_g2l(a1 + k0, &As[buf][r1 * 64 + col8]);
    cp16_g2l(w0 + k0, &Bs[buf][r0 * 64 + col8]);
    cp16_g2l(w1 + k0, &Bs[buf][r1 * 64 + col8]);
  };

  int nk = K >> 6;            // K is a multiple of 64 for every GEMM in this model
  stage(0, 0);
  for (int i = 0; i < nk; ++i) {
    if (i + 1 < nk) { stage((i + 1) & 1, (i + 1) * 64); ASYNC_WAIT(4); }
    else            { ASYNC_WAIT(0); }
    __syncthreads();
    const _Float16* as = As[i & 1];
    const _Float16* bs = Bs[i & 1];
    #pragma unroll
    for (int kc = 0; kc < 2; ++kc) {
      const _Float16* ar = as + (wm * 16 + ln) * 64 + kc * 32 + g * 16;
      v16h af = cat8(*(const v8h*)ar, *(const v8h*)(ar + 8));
      const _Float16* br0 = bs + (wn * 32 + ln) * 64 + kc * 32 + g * 16;
      const _Float16* br1 = bs + (wn * 32 + 16 + ln) * 64 + kc * 32 + g * 16;
      c0 = wmma16(af, *(const v16h*)br0, c0);
      c1 = wmma16(af, *(const v16h*)br1, c1);
    }
    __syncthreads();
  }
  #pragma unroll
  for (int j = 0; j < 2; ++j) {
    v8f c = j ? c1 : c0;
    int n = n0 + wn * 32 + j * 16 + ln;
    float bb = bias ? bias[n] : 0.f;
    #pragma unroll
    for (int r = 0; r < 8; ++r) {
      int m = m0 + wm * 16 + g * 8 + r;
      if (m >= M) continue;
      float v = c[r] + bb;
      if (act == 1) v = 0.5f * v * (1.0f + erff(v * 0.7071067811865475f));
      if (Cf) Cf[(size_t)m * N + n] = v;
      if (Ch) Ch[(size_t)m * N + n] = (_Float16)v;
    }
  }
}

// ---------------- flash attention (one wave per 16-query tile) ----------------
// Q,K,V,O layout: [B, L, 512], head h at cols h*64..h*64+63.  scale = 1/8.
__global__ __launch_bounds__(256) void flash_attn_kernel(
    const _Float16* __restrict__ Q, const _Float16* __restrict__ K,
    const _Float16* __restrict__ V, _Float16* __restrict__ O, int L) {
  __shared__ __align__(64) _Float16 sm[8 * 2560];   // per wave: P 16x32 (512) + Vt 64x32 (2048)
  int lane = threadIdx.x & 31, wave = threadIdx.x >> 5;
  int q0 = (blockIdx.x * 8 + wave) * 16;
  if (q0 >= L) return;
  int h = blockIdx.y, b = blockIdx.z;
  int g = lane >> 4, ln = lane & 15;
  _Float16* Pt = sm + wave * 2560;
  _Float16* Vt = Pt + 512;
  size_t base = (size_t)b * L;
  int qr = q0 + ln; if (qr > L - 1) qr = L - 1;
  const _Float16* qp = Q + (base + qr) * 512 + h * 64 + g * 16;
  v16h qa0 = cat8(*(const v8h*)qp,        *(const v8h*)(qp + 8));
  v16h qa1 = cat8(*(const v8h*)(qp + 32), *(const v8h*)(qp + 40));
  v8f o0 = {0,0,0,0,0,0,0,0}, o1 = o0, o2 = o0, o3 = o0;
  float mrow[8], lrow[8];
  const float NEG = -__builtin_inff();
  #pragma unroll
  for (int r = 0; r < 8; ++r) { mrow[r] = NEG; lrow[r] = 0.f; }

  for (int kt = 0; kt < L; kt += 32) {
    // prefetch next K/V tile into cache while we do softmax VALU work
    if (kt + 32 < L) {
      int pk = kt + 32 + ln;   if (pk > L - 1) pk = L - 1;
      int pv = kt + 32 + lane; if (pv > L - 1) pv = L - 1;
      __builtin_prefetch(K + (base + pk) * 512 + h * 64, 0, 1);
      __builtin_prefetch(V + (base + pv) * 512 + h * 64, 0, 1);
    }
    v8f z = {0,0,0,0,0,0,0,0};
    int k0i = kt + ln;      int kk0 = (k0i < L) ? k0i : L - 1;
    int k1i = kt + 16 + ln; int kk1 = (k1i < L) ? k1i : L - 1;
    const _Float16* kp0 = K + (base + kk0) * 512 + h * 64 + g * 16;
    const _Float16* kp1 = K + (base + kk1) * 512 + h * 64 + g * 16;
    v8f s0 = wmma16(qa0, *(const v16h*)kp0, z);
    s0 = wmma16(qa1, *(const v16h*)(kp0 + 32), s0);
    v8f s1 = wmma16(qa0, *(const v16h*)kp1, z);
    s1 = wmma16(qa1, *(const v16h*)(kp1 + 32), s1);
    bool val0 = (k0i < L), val1 = (k1i < L);
    float mnew[8];
    #pragma unroll
    for (int r = 0; r < 8; ++r) {
      float a = val0 ? s0[r] * 0.125f : NEG;
      float c = val1 ? s1[r] * 0.125f : NEG;
      s0[r] = a; s1[r] = c;
      float mx = fmaxf(a, c);
      #pragma unroll
      for (int d = 1; d < 16; d <<= 1) mx = fmaxf(mx, __shfl_xor(mx, d, 32));
      mnew[r] = fmaxf(mrow[r], mx);
    }
    #pragma unroll
    for (int r = 0; r < 8; ++r) {
      float corr = __expf(mrow[r] - mnew[r]);
      mrow[r] = mnew[r];
      lrow[r] *= corr;
      o0[r] *= corr; o1[r] *= corr; o2[r] *= corr; o3[r] *= corr;
      float p0 = __expf(s0[r] - mnew[r]);
      float p1 = __expf(s1[r] - mnew[r]);
      float rs = p0 + p1;
      #pragma unroll
      for (int d = 1; d < 16; d <<= 1) rs += __shfl_xor(rs, d, 32);
      lrow[r] += rs;
      int pr = g * 8 + r;
      Pt[pr * 32 + ln]      = (_Float16)p0;
      Pt[pr * 32 + 16 + ln] = (_Float16)p1;
    }
    // stage V chunk transposed: Vt[e][key_local]
    int kv = kt + lane; if (kv > L - 1) kv = L - 1;
    const _Float16* vp = V + (base + kv) * 512 + h * 64;
    for (int e = 0; e < 64; ++e) Vt[e * 32 + lane] = vp[e];
    // P A-fragment + 4 n-tiles of V
    const _Float16* pp = Pt + ln * 32 + g * 16;
    v16h pa = cat8(*(const v8h*)pp, *(const v8h*)(pp + 8));
    const _Float16* vb = Vt + ln * 32 + g * 16;
    o0 = wmma16(pa, *(const v16h*)(vb),           o0);
    o1 = wmma16(pa, *(const v16h*)(vb + 16 * 32), o1);
    o2 = wmma16(pa, *(const v16h*)(vb + 32 * 32), o2);
    o3 = wmma16(pa, *(const v16h*)(vb + 48 * 32), o3);
  }
  #pragma unroll
  for (int r = 0; r < 8; ++r) {
    int m = q0 + g * 8 + r;
    if (m >= L) continue;
    float inv = 1.0f / lrow[r];
    _Float16* op = O + (base + m) * 512 + h * 64 + ln;
    op[0]  = (_Float16)(o0[r] * inv);
    op[16] = (_Float16)(o1[r] * inv);
    op[32] = (_Float16)(o2[r] * inv);
    op[48] = (_Float16)(o3[r] * inv);
  }
}

// ---------------- host orchestration ----------------

static inline int cdiv(int a, int b) { return (a + b - 1) / b; }

extern "C" void kernel_launch(void* const* d_in, const int* in_sizes, int n_in,
                              void* d_out, int out_size, void* d_ws, size_t ws_size,
                              hipStream_t stream) {
  (void)in_sizes; (void)n_in; (void)out_size; (void)ws_size;
  auto IN = [&](int i) { return (const float*)d_in[i]; };

  // ---- workspace layout ----
  size_t off = 0;
  auto alloc = [&](size_t bytes) -> void* {
    void* p = (char*)d_ws + off;
    off += (bytes + 255) & ~(size_t)255;
    return p;
  };
  const size_t RMAX = (size_t)8 * 2056;     // rows with margin (max L+2 = 2050)

  _Float16 *Wq16[3], *Wk16[3], *Wv16[3], *Wo16[3], *W1_16[3], *W2_16[3];
  for (int l = 0; l < 3; ++l) {
    Wq16[l]  = (_Float16*)alloc((size_t)512 * 512 * 2);
    Wk16[l]  = (_Float16*)alloc((size_t)512 * 512 * 2);
    Wv16[l]  = (_Float16*)alloc((size_t)512 * 512 * 2);
    Wo16[l]  = (_Float16*)alloc((size_t)512 * 512 * 2);
    W1_16[l] = (_Float16*)alloc((size_t)2048 * 512 * 2);
    W2_16[l] = (_Float16*)alloc((size_t)512 * 2048 * 2);
  }
  _Float16* CW16[2];
  for (int j = 0; j < 2; ++j) CW16[j] = (_Float16*)alloc((size_t)512 * 1536 * 2);
  _Float16* FC1h = (_Float16*)alloc((size_t)2048 * 512 * 2);
  _Float16* FC2h = (_Float16*)alloc((size_t)256 * 2048 * 2);

  float*    X  = (float*)   alloc(RMAX * 512 * 4);
  _Float16* Xh = (_Float16*)alloc(RMAX * 512 * 2);
  float*    Y  = (float*)   alloc(RMAX * 512 * 4);
  char*     U1 = (char*)    alloc(RMAX * 2048 * 2);  // QKVO  |  FFN hidden  |  im2col
  _Float16* Qh = (_Float16*)U1;
  _Float16* Kh = Qh + RMAX * 512;
  _Float16* Vh = Kh + RMAX * 512;
  _Float16* Oh = Vh + RMAX * 512;
  _Float16* Hh = (_Float16*)U1;
  _Float16* XC = (_Float16*)U1;
  _Float16* inph  = (_Float16*)alloc(16 * 512 * 2);
  float*    Hhead = (float*)   alloc(16 * 2048 * 4);
  _Float16* H1h   = (_Float16*)alloc(16 * 2048 * 2);

  auto conv16 = [&](const float* s, _Float16* d, size_t n) {
    f32_to_f16_kernel<<<dim3((unsigned)((n + 255) / 256)), 256, 0, stream>>>(s, d, n);
  };

  // ---- weight conversion (inside graph; deterministic) ----
  for (int l = 0; l < 3; ++l) {
    int bse = 3 + l * 16;
    conv16(IN(bse + 0),  Wq16[l],  (size_t)512 * 512);
    conv16(IN(bse + 2),  Wk16[l],  (size_t)512 * 512);
    conv16(IN(bse + 4),  Wv16[l],  (size_t)512 * 512);
    conv16(IN(bse + 6),  Wo16[l],  (size_t)512 * 512);
    conv16(IN(bse + 8),  W1_16[l], (size_t)2048 * 512);
    conv16(IN(bse + 10), W2_16[l], (size_t)512 * 2048);
  }
  for (int j = 0; j < 2; ++j)
    conv_w_kernel<<<cdiv(512 * 1536, 256), 256, 0, stream>>>(IN(51 + j * 4), CW16[j]);
  conv16(IN(61), FC1h, (size_t)2048 * 512);
  conv16(IN(63), FC2h, (size_t)256 * 2048);

  // ---- token embedding + positional encoding ----
  int L = 2048;
  tok_pos_kernel<<<8 * L, 128, 0, stream>>>(IN(0), IN(1), IN(2), X, Xh, L);

  // ---- encoder layers ----
  for (int li = 0; li < 3; ++li) {
    int bse = 3 + li * 16;
    int M = 8 * L;
    dim3 g512(cdiv(M, 64), 8);
    gemm_kernel<<<g512, 256, 0, stream>>>(Xh, Wq16[li], IN(bse + 1), nullptr, Qh, M, 512, 512, 0);
    gemm_kernel<<<g512, 256, 0, stream>>>(Xh, Wk16[li], IN(bse + 3), nullptr, Kh, M, 512, 512, 0);
    gemm_kernel<<<g512, 256, 0, stream>>>(Xh, Wv16[li], IN(bse + 5), nullptr, Vh, M, 512, 512, 0);
    flash_attn_kernel<<<dim3(cdiv(L, 128), 8, 8), 256, 0, stream>>>(Qh, Kh, Vh, Oh, L);
    gemm_kernel<<<g512, 256, 0, stream>>>(Oh, Wo16[li], IN(bse + 7), Y, nullptr, M, 512, 512, 0);
    ln_kernel<<<M, 128, 0, stream>>>(X, Y, IN(bse + 12), IN(bse + 13), X, Xh);
    // FFN
    gemm_kernel<<<dim3(cdiv(M, 64), 32), 256, 0, stream>>>(Xh, W1_16[li], IN(bse + 9), nullptr, Hh, M, 2048, 512, 1);
    gemm_kernel<<<dim3(cdiv(M, 64), 8), 256, 0, stream>>>(Hh, W2_16[li], IN(bse + 11), Y, nullptr, M, 512, 2048, 0);
    ln_kernel<<<M, 128, 0, stream>>>(X, Y, IN(bse + 14), IN(bse + 15), X, Xh);
    // distil
    if (li < 2) {
      int db = 51 + li * 4;
      int Lc = L + 2;
      im2col_kernel<<<8 * Lc, 256, 0, stream>>>(Xh, XC, L, Lc);
      gemm_kernel<<<dim3(cdiv(8 * Lc, 64), 8), 256, 0, stream>>>(XC, CW16[li], IN(db + 1), Y, nullptr, 8 * Lc, 512, 1536, 0);
      int Lo = (Lc - 1) / 2 + 1;
      distil_post_kernel<<<8 * Lo, 256, 0, stream>>>(Y, IN(db + 2), IN(db + 3), X, Xh, Lc, Lo);
      L = Lo;
    }
  }

  // ---- final norm + head ----
  ln_kernel<<<8 * L, 128, 0, stream>>>(X, nullptr, IN(59), IN(60), X, Xh);
  take_last_kernel<<<cdiv(8 * 512, 256), 256, 0, stream>>>(Xh, inph, L);
  gemm_kernel<<<dim3(1, 32), 256, 0, stream>>>(inph, FC1h, IN(62), Hhead, nullptr, 8, 2048, 512, 0);
  bnelu_kernel<<<cdiv(8 * 2048, 256), 256, 0, stream>>>(Hhead, IN(65), IN(66), H1h, 8 * 2048, 2048);
  gemm_kernel<<<dim3(1, 4), 256, 0, stream>>>(H1h, FC2h, IN(64), (float*)d_out, nullptr, 8, 256, 2048, 0);
}
